// AACls_87282325389858
// MI455X (gfx1250) — matrix operand order repaired and checked by reading, not compiled, not run on hardware
//
#include <hip/hip_runtime.h>
#include <hip/hip_bf16.h>
#include <math.h>

// ---------------------------------------------------------------------------
// Types for CDNA5 WMMA (wave32) and TDM descriptors
// ---------------------------------------------------------------------------
typedef __attribute__((ext_vector_type(16))) __bf16 v16bf;
typedef __attribute__((ext_vector_type(8)))  float  v8f;
typedef unsigned int u32x4 __attribute__((ext_vector_type(4)));
typedef int          i32x8 __attribute__((ext_vector_type(8)));
typedef int          i32x4 __attribute__((ext_vector_type(4)));

__device__ __forceinline__ unsigned short f2b(float f) {
    unsigned int u = __float_as_uint(f);
    u += 0x7fffu + ((u >> 16) & 1u);          // round-to-nearest-even
    return (unsigned short)(u >> 16);
}
__device__ __forceinline__ float b2f(unsigned short h) {
    return __uint_as_float(((unsigned int)h) << 16);
}

union ABu {
    uint4          u[2];
    unsigned short s[16];
    v16bf          v;
};

// ---------------------------------------------------------------------------
// TDM-staged bf16 WMMA GEMM:  Out[M,N] = A[M,K] * Bw[N,K]^T  (row-major)
// Block = 128 threads (4 waves) -> 64 rows x 64 cols of output.
// Per K-phase (KC=768): wave0 issues one TENSOR_LOAD_TO_LDS moving the
// 64-row x 768-col B panel (96KB + pad) into LDS, waits TENSORcnt, barrier;
// all 4 waves then run the 24-step WMMA K-loop reading B from LDS
// (ds_load_b128) with register-double-buffered A from global.
// TDM LDS padding: 16B per 512B interval -> row pitch 1584B (bank spread).
// EPI: 0=store bf16, 1=2*(acc+bias) f32, 2=gelu(acc+bias) bf16,
//      3=acc+bias+res f32
// ---------------------------------------------------------------------------
#define BPITCH 1584   // 1536 data + 3*16 pad bytes per staged B row

template <int EPI>
__global__ __launch_bounds__(128) void wmma_gemm_tdm(
    const unsigned short* __restrict__ A, int lda,
    const unsigned short* __restrict__ Bw, int ldb,
    void* __restrict__ Outp, int ldc,
    const float* __restrict__ aux,
    const float* __restrict__ res, int ldres,
    int K)
{
    constexpr int KC = 768;
    __shared__ unsigned short lb[(64 * BPITCH) / 2];

    const int tid  = threadIdx.x;
    const int wave = tid >> 5;
    const int lane = tid & 31;
    const int half = lane >> 4;
    const int l16  = lane & 15;
    const int rowBase = blockIdx.x * 64 + wave * 16;
    const int colBase = blockIdx.y * 64;

    v8f acc[4];
#pragma unroll
    for (int j = 0; j < 4; ++j)
#pragma unroll
        for (int r = 0; r < 8; ++r) acc[j][r] = 0.f;

    const char* lbb = (const char*)lb;

    for (int p = 0; p < K; p += KC) {
        if (p) __syncthreads();           // all waves done reading prev panel
        if (wave == 0) {
            // Build D# for a 2D tile: tile_dim0=KC elems (2B), tile_dim1=64
            const unsigned long long ga =
                (unsigned long long)(const void*)(Bw + (size_t)colBase * ldb + p);
            const unsigned lds_off =
                (unsigned)(unsigned long long)(void*)lb;   // LDS byte address: low 32 bits of lb
            u32x4 g0;
            g0[0] = 1u;                                    // count=1, user desc
            g0[1] = lds_off;                               // lds_addr (bytes)
            g0[2] = (unsigned)ga;                          // global_addr lo
            g0[3] = (unsigned)(ga >> 32) | (2u << 30);     // global_addr hi | type=2
            i32x8 g1;
            g1[0] = (int)((1u << 16)                       // data_size = 2 bytes
                        | (1u << 20)                       // pad_enable
                        | (6u << 22)                       // pad_interval: 128 DW = 512B
                        | (3u << 25));                     // pad_amount: 4 DW = 16B
            g1[1] = (int)((unsigned)(KC & 0xffff) << 16);  // tensor_dim0[15:0]
            g1[2] = (int)(64u << 16);                      // dim0 hi=0 | tensor_dim1[15:0]
            g1[3] = (int)((unsigned)KC << 16);             // dim1 hi=0 | tile_dim0
            g1[4] = 64;                                    // tile_dim1 | tile_dim2=0
            g1[5] = (int)(unsigned)ldb;                    // tensor_dim0_stride lo
            g1[6] = 0;
            g1[7] = 0;
            i32x4 g2 = {0, 0, 0, 0};
            i32x4 g3 = {0, 0, 0, 0};
            i32x8 g4 = {0, 0, 0, 0, 0, 0, 0, 0};           // 6-arg toolchain extra group
            __builtin_amdgcn_tensor_load_to_lds(g0, g1, g2, g3, g4, 0);
            __builtin_amdgcn_s_wait_tensorcnt(0);
        }
        __syncthreads();

        const unsigned short* ar = A + (size_t)(rowBase + l16) * lda + p;
        ABu a, an;
        an.u[0] = *(const uint4*)(ar + half * 16);
        an.u[1] = *(const uint4*)(ar + half * 16 + 8);
        for (int k0 = 0; k0 < KC; k0 += 32) {
            a = an;
            if (k0 + 32 < KC) {                            // prefetch next A chunk
                an.u[0] = *(const uint4*)(ar + k0 + 32 + half * 16);
                an.u[1] = *(const uint4*)(ar + k0 + 40 + half * 16);
            }
            const int o     = (k0 + half * 16) * 2;        // byte offset in B row
            const int obase = o + ((o >> 9) << 4);         // + TDM pad skips
#pragma unroll
            for (int j = 0; j < 4; ++j) {
                const char* br = lbb + (j * 16 + l16) * BPITCH + obase;
                ABu bt;
                bt.u[0] = *(const uint4*)(br);
                bt.u[1] = *(const uint4*)(br + 16);
                acc[j] = __builtin_amdgcn_wmma_f32_16x16x32_bf16(
                    false, a.v, false, bt.v, (short)0, acc[j], false, false);
            }
        }
    }

    float*          outF = (float*)Outp;
    unsigned short* outB = (unsigned short*)Outp;
#pragma unroll
    for (int j = 0; j < 4; ++j) {
        const int col = colBase + j * 16 + l16;
#pragma unroll
        for (int r = 0; r < 8; ++r) {
            const int row = rowBase + r + half * 8;
            float v = acc[j][r];
            if (EPI == 0) {
                outB[(size_t)row * ldc + col] = f2b(v);
            } else if (EPI == 1) {
                v = 2.f * (v + aux[col]);
                outF[(size_t)row * ldc + col] = v;
            } else if (EPI == 2) {
                v += aux[col];
                v = 0.5f * v * (1.f + erff(v * 0.7071067811865475f));
                outB[(size_t)row * ldc + col] = f2b(v);
            } else if (EPI == 3) {
                v += aux[col] + res[(size_t)row * ldres + col];
                outF[(size_t)row * ldc + col] = v;
            }
        }
    }
}

// ---------------------------------------------------------------------------
// Register-direct bf16 WMMA GEMM (batched over z), used for attention:
// EPI 4 = acc*scale -> f32 (QK^T), EPI 5 = acc*gate(row) -> bf16 (attn@V,
// with AF32: A read as f32 and converted to bf16 on the fly).
// ---------------------------------------------------------------------------
template <int EPI, int NT, bool AF32>
__global__ __launch_bounds__(32) void wmma_gemm_reg(
    const void* __restrict__ Ap, int lda,
    const unsigned short* __restrict__ Bw, int ldb,
    void* __restrict__ Outp, int ldc,
    const float* __restrict__ aux, int auxStride,
    int K, float scale, int Hdiv,
    long long aZd, long long aZm, long long bZd, long long bZm,
    long long cZd, long long cZm, long long xZd, long long xZm)
{
    const int lane = threadIdx.x & 31;
    const int half = lane >> 4;
    const int l16  = lane & 15;
    const int rowBase = blockIdx.x * 16;
    const int colBase = blockIdx.y * (NT * 16);
    const long long zd = (long long)(blockIdx.z / Hdiv);
    const long long zm = (long long)(blockIdx.z % Hdiv);

    const unsigned short* Ab = (const unsigned short*)Ap + (AF32 ? 0 : (zd * aZd + zm * aZm));
    const float*          Af = (const float*)Ap          + (AF32 ? (zd * aZd + zm * aZm) : 0);
    const unsigned short* B  = Bw + zd * bZd + zm * bZm;

    v8f acc[NT];
#pragma unroll
    for (int j = 0; j < NT; ++j)
#pragma unroll
        for (int r = 0; r < 8; ++r) acc[j][r] = 0.f;

    for (int k0 = 0; k0 < K; k0 += 32) {
        ABu a;
        if (!AF32) {
            const unsigned short* ar = Ab + (size_t)(rowBase + l16) * lda;
            a.u[0] = *(const uint4*)(ar + k0 + half * 8);
            a.u[1] = *(const uint4*)(ar + k0 + 16 + half * 8);
        } else {
            const float* ar = Af + (size_t)(rowBase + l16) * lda;
            const float4 f0 = *(const float4*)(ar + k0 + half * 8);
            const float4 f1 = *(const float4*)(ar + k0 + half * 8 + 4);
            const float4 f2 = *(const float4*)(ar + k0 + 16 + half * 8);
            const float4 f3 = *(const float4*)(ar + k0 + 16 + half * 8 + 4);
            a.s[0]  = f2b(f0.x); a.s[1]  = f2b(f0.y); a.s[2]  = f2b(f0.z); a.s[3]  = f2b(f0.w);
            a.s[4]  = f2b(f1.x); a.s[5]  = f2b(f1.y); a.s[6]  = f2b(f1.z); a.s[7]  = f2b(f1.w);
            a.s[8]  = f2b(f2.x); a.s[9]  = f2b(f2.y); a.s[10] = f2b(f2.z); a.s[11] = f2b(f2.w);
            a.s[12] = f2b(f3.x); a.s[13] = f2b(f3.y); a.s[14] = f2b(f3.z); a.s[15] = f2b(f3.w);
        }
#pragma unroll
        for (int j = 0; j < NT; ++j) {
            const unsigned short* br = B + (size_t)(colBase + j * 16 + l16) * ldb + k0 + half * 8;
            ABu bt;
            bt.u[0] = *(const uint4*)(br);
            bt.u[1] = *(const uint4*)(br + 16);
            acc[j] = __builtin_amdgcn_wmma_f32_16x16x32_bf16(
                false, a.v, false, bt.v, (short)0, acc[j], false, false);
        }
    }

    float*          outF = (float*)Outp          + cZd * zd + cZm * zm;
    unsigned short* outB = (unsigned short*)Outp + cZd * zd + cZm * zm;
    const float*    auxp = aux ? (aux + xZd * zd + xZm * zm) : nullptr;

#pragma unroll
    for (int j = 0; j < NT; ++j) {
        const int col = colBase + j * 16 + l16;
#pragma unroll
        for (int r = 0; r < 8; ++r) {
            const int row = rowBase + r + half * 8;
            float v = acc[j][r];
            if (EPI == 4) {
                outF[(size_t)row * ldc + col] = v * scale;
            } else if (EPI == 5) {
                v *= auxp[(size_t)row * auxStride];
                outB[(size_t)row * ldc + col] = f2b(v);
            }
        }
    }
}

// ---------------------------------------------------------------------------
// LayerNorm over C per row: f32 in -> bf16 out
// ---------------------------------------------------------------------------
__global__ __launch_bounds__(256) void ln_kernel(
    const float* __restrict__ x, const float* __restrict__ w,
    const float* __restrict__ b, unsigned short* __restrict__ out, int C)
{
    __shared__ float ssum[256], ssq[256];
    const int tid = threadIdx.x;
    const size_t row = blockIdx.x;
    const float* xr = x + row * C;
    float s = 0.f, q = 0.f;
    for (int c = tid; c < C; c += 256) { float v = xr[c]; s += v; q += v * v; }
    ssum[tid] = s; ssq[tid] = q;
    __syncthreads();
    for (int st = 128; st > 0; st >>= 1) {
        if (tid < st) { ssum[tid] += ssum[tid + st]; ssq[tid] += ssq[tid + st]; }
        __syncthreads();
    }
    const float mu  = ssum[0] / (float)C;
    const float var = ssq[0] / (float)C - mu * mu;
    const float rs  = rsqrtf(var + 1e-5f);
    for (int c = tid; c < C; c += 256)
        out[row * C + c] = f2b((xr[c] - mu) * rs * w[c] + b[c]);
}

// ---------------------------------------------------------------------------
// Row softmax, in place, f32 (rows of length M=1024)
// ---------------------------------------------------------------------------
__global__ __launch_bounds__(256) void softmax_kernel(float* __restrict__ p, int M)
{
    __shared__ float red[256];
    const int tid = threadIdx.x;
    float* row = p + (size_t)blockIdx.x * M;
    float m = -3.4e38f;
    for (int c = tid; c < M; c += 256) m = fmaxf(m, row[c]);
    red[tid] = m; __syncthreads();
    for (int st = 128; st > 0; st >>= 1) {
        if (tid < st) red[tid] = fmaxf(red[tid], red[tid + st]);
        __syncthreads();
    }
    const float mx = red[0];
    __syncthreads();
    float s = 0.f;
    for (int c = tid; c < M; c += 256) { float e = expf(row[c] - mx); row[c] = e; s += e; }
    red[tid] = s; __syncthreads();
    for (int st = 128; st > 0; st >>= 1) {
        if (tid < st) red[tid] += red[tid + st];
        __syncthreads();
    }
    const float inv = 1.f / red[0];
    for (int c = tid; c < M; c += 256) row[c] *= inv;
}

// ---------------------------------------------------------------------------
// Gating: one wave per token. top_k(gates, 6) over 6 gates keeps all ->
// rg = 6*softmax(logits); sg = 2*softmax(wg1); w0 = 2*softmax(wg0);
// mg = [w0_0*sg(2), w0_1*rg(6)]  (H=8 values/token)
// ---------------------------------------------------------------------------
__global__ __launch_bounds__(256) void gate_kernel(
    const unsigned short* __restrict__ xn, const float* __restrict__ wg,
    const float* __restrict__ wg0, const float* __restrict__ wg1,
    float* __restrict__ mg)
{
    const int wave = threadIdx.x >> 5;
    const int lane = threadIdx.x & 31;
    const int token = blockIdx.x * 8 + wave;
    const unsigned short* xr = xn + (size_t)token * 768;
    float acc[10];
#pragma unroll
    for (int i = 0; i < 10; ++i) acc[i] = 0.f;
    for (int c = lane; c < 768; c += 32) {
        const float xv = b2f(xr[c]);
#pragma unroll
        for (int r = 0; r < 6; ++r) acc[r] += xv * wg[r * 768 + c];
        acc[6] += xv * wg1[c];
        acc[7] += xv * wg1[768 + c];
        acc[8] += xv * wg0[c];
        acc[9] += xv * wg0[768 + c];
    }
#pragma unroll
    for (int m = 16; m > 0; m >>= 1)
#pragma unroll
        for (int i = 0; i < 10; ++i) acc[i] += __shfl_xor(acc[i], m, 32);
    if (lane == 0) {
        float rmx = acc[0];
#pragma unroll
        for (int i = 1; i < 6; ++i) rmx = fmaxf(rmx, acc[i]);
        float rs = 0.f, re[6];
#pragma unroll
        for (int i = 0; i < 6; ++i) { re[i] = expf(acc[i] - rmx); rs += re[i]; }
        const float smx = fmaxf(acc[6], acc[7]);
        const float s0 = expf(acc[6] - smx), s1 = expf(acc[7] - smx);
        const float wmx = fmaxf(acc[8], acc[9]);
        const float w0e = expf(acc[8] - wmx), w1e = expf(acc[9] - wmx);
        const float w00 = 2.f * w0e / (w0e + w1e);
        const float w01 = 2.f * w1e / (w0e + w1e);
        float* o = mg + (size_t)token * 8;
        o[0] = w00 * 2.f * s0 / (s0 + s1);
        o[1] = w00 * 2.f * s1 / (s0 + s1);
#pragma unroll
        for (int i = 0; i < 6; ++i) o[2 + i] = w01 * 6.f * re[i] / rs;
    }
}

// ---------------------------------------------------------------------------
// Transpose V slice of qkv into Vt[b,h,d,m] (bf16) so B-operand is contiguous
// ---------------------------------------------------------------------------
__global__ __launch_bounds__(256) void vt_kernel(
    const unsigned short* __restrict__ qkv, unsigned short* __restrict__ vt, int total)
{
    const int idx = blockIdx.x * 256 + threadIdx.x;
    if (idx >= total) return;
    const int m = idx & 1023;
    const int d = (idx >> 10) % 96;
    const int z = idx / (96 * 1024);
    const int b = z >> 3, h = z & 7;
    vt[idx] = qkv[(size_t)(b * 1024 + m) * 2304 + (16 + h) * 96 + d];
}

// ---------------------------------------------------------------------------
// f32 -> bf16 elementwise
// ---------------------------------------------------------------------------
__global__ __launch_bounds__(256) void cvt_kernel(
    const float* __restrict__ in, unsigned short* __restrict__ out, int n)
{
    for (int i = blockIdx.x * 256 + threadIdx.x; i < n; i += gridDim.x * 256)
        out[i] = f2b(in[i]);
}

// ---------------------------------------------------------------------------
extern "C" void kernel_launch(void* const* d_in, const int* in_sizes, int n_in,
                              void* d_out, int out_size, void* d_ws, size_t ws_size,
                              hipStream_t stream)
{
    (void)in_sizes; (void)n_in; (void)out_size; (void)ws_size;
    const float* x      = (const float*)d_in[0];
    const float* n1w    = (const float*)d_in[1];
    const float* n1b    = (const float*)d_in[2];
    const float* qkv_w  = (const float*)d_in[3];
    const float* wg_w   = (const float*)d_in[4];
    const float* wg0_w  = (const float*)d_in[5];
    const float* wg1_w  = (const float*)d_in[6];
    const float* proj_w = (const float*)d_in[7];
    const float* proj_b = (const float*)d_in[8];
    const float* n2w    = (const float*)d_in[9];
    const float* n2b    = (const float*)d_in[10];
    const float* fc1_w  = (const float*)d_in[11];
    const float* fc1_b  = (const float*)d_in[12];
    const float* fc2_w  = (const float*)d_in[13];
    const float* fc2_b  = (const float*)d_in[14];

    constexpr int Bb = 8, Nn = 1024, Cc = 768, Hh = 8, Dd = 96, HID = 3072;
    constexpr int BN = Bb * Nn;
    constexpr int QKVN = 3 * Hh * Dd; // 2304

    char* ws = (char*)d_ws;
    size_t off = 0;
    auto alloc = [&](size_t bytes) -> void* {
        void* p = ws + off;
        off += (bytes + 255) & ~(size_t)255;
        return p;
    };
    unsigned short* xn    = (unsigned short*)alloc((size_t)BN * Cc * 2);
    unsigned short* qkvb  = (unsigned short*)alloc((size_t)BN * QKVN * 2);
    unsigned short* vt    = (unsigned short*)alloc((size_t)Bb * Hh * Dd * Nn * 2);
    unsigned short* og    = (unsigned short*)alloc((size_t)BN * Cc * 2);
    unsigned short* hn    = (unsigned short*)alloc((size_t)BN * Cc * 2);
    unsigned short* hid   = (unsigned short*)alloc((size_t)BN * HID * 2);
    float*          hbuf  = (float*)alloc((size_t)BN * Cc * 4);
    float*          mg    = (float*)alloc((size_t)BN * Hh * 4);
    unsigned short* wqkv  = (unsigned short*)alloc((size_t)QKVN * Cc * 2);
    unsigned short* wproj = (unsigned short*)alloc((size_t)Cc * Cc * 2);
    unsigned short* wfc1  = (unsigned short*)alloc((size_t)HID * Cc * 2);
    unsigned short* wfc2  = (unsigned short*)alloc((size_t)Cc * HID * 2);

    float* out1 = (float*)d_out;                     // [B,N,C] h+mlp
    float* attn = out1 + (size_t)BN * Cc;            // [B,H,N,N] f32

    // Weight conversions to bf16
    cvt_kernel<<<dim3((QKVN * Cc + 255) / 256), 256, 0, stream>>>(qkv_w, wqkv, QKVN * Cc);
    cvt_kernel<<<dim3((Cc * Cc + 255) / 256), 256, 0, stream>>>(proj_w, wproj, Cc * Cc);
    cvt_kernel<<<dim3((HID * Cc + 255) / 256), 256, 0, stream>>>(fc1_w, wfc1, HID * Cc);
    cvt_kernel<<<dim3((Cc * HID + 255) / 256), 256, 0, stream>>>(fc2_w, wfc2, Cc * HID);

    // LN1 -> xn (bf16)
    ln_kernel<<<dim3(BN), 256, 0, stream>>>(x, n1w, n1b, xn, Cc);

    // Gating -> mg [BN, 8]
    gate_kernel<<<dim3(BN / 8), 256, 0, stream>>>(xn, wg_w, wg0_w, wg1_w, mg);

    // QKV GEMM (TDM-staged): [BN,2304] = xn @ qkv_w^T  (bf16 out)
    wmma_gemm_tdm<0><<<dim3(BN / 64, QKVN / 64), 128, 0, stream>>>(
        xn, Cc, wqkv, Cc, qkvb, QKVN, nullptr, nullptr, 0, Cc);

    // V transpose -> Vt[b,h,d,m]
    vt_kernel<<<dim3((Bb * Hh * Dd * Nn + 255) / 256), 256, 0, stream>>>(
        qkvb, vt, Bb * Hh * Dd * Nn);

    // S = scale * Q @ K^T  -> attn region (f32), batched over z = b*H + h
    const float scl = 0.1020620726159658f; // 96^-0.5
    wmma_gemm_reg<4, 4, false><<<dim3(Nn / 16, Nn / 64, Bb * Hh), 32, 0, stream>>>(
        qkvb, QKVN, qkvb + Hh * Dd, QKVN, attn, Nn, nullptr, 0,
        Dd, scl, Hh,
        (long long)Nn * QKVN, (long long)Dd,
        (long long)Nn * QKVN, (long long)Dd,
        (long long)Hh * Nn * Nn, (long long)Nn * Nn, 0, 0);

    // softmax rows in place (B*H*N rows of length N)
    softmax_kernel<<<dim3(Bb * Hh * Nn), 256, 0, stream>>>(attn, Nn);

    // O = attn @ V, gated per (token, head) -> og[BN, 768] bf16
    wmma_gemm_reg<5, 3, true><<<dim3(Nn / 16, Dd / 48, Bb * Hh), 32, 0, stream>>>(
        attn, Nn, vt, Nn, og, Cc, mg, Hh,
        Nn, 1.f, Hh,
        (long long)Hh * Nn * Nn, (long long)Nn * Nn,
        (long long)Hh * Dd * Nn, (long long)Dd * Nn,
        (long long)Nn * Cc, (long long)Dd,
        (long long)Nn * Hh, 1LL);

    // h = 2 * (og @ proj_w^T + proj_b)  -> hbuf (f32)  (TDM-staged)
    wmma_gemm_tdm<1><<<dim3(BN / 64, Cc / 64), 128, 0, stream>>>(
        og, Cc, wproj, Cc, hbuf, Cc, proj_b, nullptr, 0, Cc);

    // LN2 -> hn (bf16)
    ln_kernel<<<dim3(BN), 256, 0, stream>>>(hbuf, n2w, n2b, hn, Cc);

    // FC1 + exact GELU -> hid (bf16)  (TDM-staged)
    wmma_gemm_tdm<2><<<dim3(BN / 64, HID / 64), 128, 0, stream>>>(
        hn, Cc, wfc1, Cc, hid, HID, fc1_b, nullptr, 0, Cc);

    // out1 = hid @ fc2_w^T + fc2_b + h  (f32 -> d_out)  (TDM-staged, 4 K-phases)
    wmma_gemm_tdm<3><<<dim3(BN / 64, Cc / 64), 128, 0, stream>>>(
        hid, HID, wfc2, HID, out1, Cc, fc2_b, hbuf, Cc, HID);
}
